// DLRM_Net_59012850647143
// MI455X (gfx1250) — matrix operand order repaired and checked by reading, not compiled, not run on hardware
//
#include <hip/hip_runtime.h>
#include <hip/hip_bf16.h>

// ---------------------------------------------------------------------------
// DLRM forward for gfx1250 (MI455X): bf16 WMMA GEMMs + fused gather/interact.
// ---------------------------------------------------------------------------

typedef __bf16 bf16;
typedef __attribute__((ext_vector_type(8)))  __bf16 v8bf;
typedef __attribute__((ext_vector_type(16))) __bf16 v16bf;
typedef __attribute__((ext_vector_type(8)))  float  v8f;
typedef __attribute__((ext_vector_type(2)))  float  v2f;

#define DLRM_B    4096
#define DLRM_T    26
#define DLRM_L    4
#define DLRM_ROWS 100000
#define DLRM_M    64

__device__ __forceinline__ v16bf cat16(v8bf lo, v8bf hi) {
  return __builtin_shufflevector(lo, hi, 0,1,2,3,4,5,6,7,8,9,10,11,12,13,14,15);
}

// A-fragment (16x32 bf16, MxK) per CDNA5 layout:
// lane l: M = l&15; vector elem i -> K = kb + (i/8)*16 + (l>>4)*8 + (i%8)
// => two contiguous 8-element (16B) chunks at kb + half*8 and kb + 16 + half*8.
__device__ __forceinline__ v16bf load_a_frag(const bf16* A, int ld, int rowBase,
                                             int kb, int lane) {
  const int half = lane >> 4, mr = lane & 15;
  const bf16* p = A + (size_t)(rowBase + mr) * ld + kb + half * 8;
  v8bf lo = *(const v8bf*)p;
  v8bf hi = *(const v8bf*)(p + 16);
  return cat16(lo, hi);
}

// B-fragment (32x16 bf16, KxN) with B[k][n] = W[n][k] (W row-major [N,Kp]):
// lane l: N = l&15; vector elem i -> K = kb + (i/8)*16 + (l>>4)*8 + (i%8).
__device__ __forceinline__ v16bf load_b_frag(const bf16* W, int ld, int colBase,
                                             int kb, int lane) {
  const int half = lane >> 4, nc = lane & 15;
  const bf16* p = W + (size_t)(colBase + nc) * ld + kb + half * 8;
  v8bf lo = *(const v8bf*)p;
  v8bf hi = *(const v8bf*)(p + 16);
  return cat16(lo, hi);
}

__device__ __forceinline__ v8f wmma_bf16(v16bf a, v16bf b, v8f c) {
  return __builtin_amdgcn_wmma_f32_16x16x32_bf16(false, a, false, b,
                                                 (short)0, c, false, false);
}

// ---------------------------------------------------------------------------
// fp32 -> bf16 convert with zero padding: in [N,K] -> out [Np,Kp]
// ---------------------------------------------------------------------------
__global__ void cvt_pad_bf16(const float* __restrict__ in, bf16* __restrict__ out,
                             int N, int K, int Np, int Kp) {
  int i = blockIdx.x * blockDim.x + threadIdx.x;
  int total = Np * Kp;
  if (i >= total) return;
  int n = i / Kp, k = i - n * Kp;
  float v = (n < N && k < K) ? in[(size_t)n * K + k] : 0.f;
  out[i] = (bf16)v;
}

// ---------------------------------------------------------------------------
// GEMM: out[M,N] = act(A[M,Kp] * W[N,Kp]^T + bias). One wave -> 16x(16*NSUB).
// ---------------------------------------------------------------------------
template <int NSUB, bool RELU>
__global__ void __launch_bounds__(256)
gemm_bias_act(const bf16* __restrict__ A, const bf16* __restrict__ W,
              const float* __restrict__ bias, bf16* __restrict__ out,
              int M, int N, int Kp) {
  const int lane = threadIdx.x & 31;
  const int g = (blockIdx.x * blockDim.x + threadIdx.x) >> 5;
  const int tilesM = M >> 4;
  const int tilesNg = N / (16 * NSUB);
  if (g >= tilesM * tilesNg) return;
  const int tm  = g % tilesM;
  const int tn0 = (g / tilesM) * NSUB;

  const v8f zero = {0.f,0.f,0.f,0.f,0.f,0.f,0.f,0.f};
  v8f c[NSUB];
#pragma unroll
  for (int s = 0; s < NSUB; ++s) c[s] = zero;

  for (int kb = 0; kb < Kp; kb += 32) {
    v16bf a = load_a_frag(A, Kp, tm * 16, kb, lane);
#pragma unroll
    for (int s = 0; s < NSUB; ++s) {
      v16bf b = load_b_frag(W, Kp, (tn0 + s) * 16, kb, lane);
      c[s] = wmma_bf16(a, b, c[s]);
    }
  }

  const int half = lane >> 4, nc = lane & 15;
#pragma unroll
  for (int s = 0; s < NSUB; ++s) {
    const int n = (tn0 + s) * 16 + nc;
    const float bn = bias[n];
#pragma unroll
    for (int r = 0; r < 8; ++r) {
      const int m = tm * 16 + half * 8 + r;      // C/D layout: M = r + 8*(l>=16)
      float v = c[s][r] + bn;
      if (RELU) v = fmaxf(v, 0.f);
      out[(size_t)m * N + n] = (bf16)v;
    }
  }
}

// ---------------------------------------------------------------------------
// Final layer: [M,256] x [16,256]^T (only n=0 real) + bias, sigmoid -> f32.
// ---------------------------------------------------------------------------
__global__ void __launch_bounds__(256)
gemm_final_sigmoid(const bf16* __restrict__ A, const bf16* __restrict__ W,
                   const float* __restrict__ bias, float* __restrict__ out,
                   int M, int Kp) {
  const int lane = threadIdx.x & 31;
  const int g = (blockIdx.x * blockDim.x + threadIdx.x) >> 5;
  const int tilesM = M >> 4;
  if (g >= tilesM) return;

  v8f c = {0.f,0.f,0.f,0.f,0.f,0.f,0.f,0.f};
  for (int kb = 0; kb < Kp; kb += 32) {
    v16bf a = load_a_frag(A, Kp, g * 16, kb, lane);
    v16bf b = load_b_frag(W, Kp, 0, kb, lane);
    c = wmma_bf16(a, b, c);
  }
  const int half = lane >> 4, nc = lane & 15;
  if (nc == 0) {
    const float b0 = bias[0];
#pragma unroll
    for (int r = 0; r < 8; ++r) {
      const int m = g * 16 + half * 8 + r;
      const float v = c[r] + b0;
      out[m] = 1.f / (1.f + __expf(-v));
    }
  }
}

// ---------------------------------------------------------------------------
// Fused embedding gather + sum + pairwise interaction.
// One wave per sample: Tcat[32x64] built in LDS (rows: bottom-MLP x, 26 summed
// embeddings, 5 zero pad), Z = Tcat*Tcat^T via 3 WMMA tiles, strict lower
// triangle scattered into R[b, 64..414]; R[b,0..63]=x; R[b,415]=0 (pad).
// ---------------------------------------------------------------------------
__global__ void __launch_bounds__(256)
interact_kernel(const bf16* __restrict__ h3, const float* __restrict__ emb,
                const int* __restrict__ sidx, bf16* __restrict__ R) {
  __shared__ bf16 tc[8][32 * 64];                 // 32 KB: 8 samples/block
  const int lane = threadIdx.x & 31;
  const int w = threadIdx.x >> 5;
  const int b = blockIdx.x * 8 + w;
  bf16* T = &tc[w][0];

  // Row 0: bottom-MLP output (already bf16)
  T[2 * lane]     = h3[(size_t)b * 64 + 2 * lane];
  T[2 * lane + 1] = h3[(size_t)b * 64 + 2 * lane + 1];

  // Rows 1..26: sum of 4 gathered embedding rows, 2 columns per lane
  for (int t = 0; t < DLRM_T; ++t) {
    float ax = 0.f, ay = 0.f;
    const int* ip = sidx + ((size_t)b * DLRM_T + t) * DLRM_L;
#pragma unroll
    for (int j = 0; j < DLRM_L; ++j) {
      const int row = ip[j];
      const float* p = emb + ((size_t)t * DLRM_ROWS + (size_t)row) * DLRM_M + 2 * lane;
      v2f v = *(const v2f*)p;
      ax += v.x; ay += v.y;
    }
    T[(1 + t) * 64 + 2 * lane]     = (bf16)ax;
    T[(1 + t) * 64 + 2 * lane + 1] = (bf16)ay;
  }
  // Rows 27..31: zero pad
#pragma unroll
  for (int r = 27; r < 32; ++r) {
    T[r * 64 + 2 * lane]     = (bf16)0.f;
    T[r * 64 + 2 * lane + 1] = (bf16)0.f;
  }
  __syncthreads();

  // Z = Tcat * Tcat^T, tiles (0,0),(1,0),(1,1); (0,1) is upper-tri only.
  const v8f zero = {0.f,0.f,0.f,0.f,0.f,0.f,0.f,0.f};
  v8f c00 = zero, c10 = zero, c11 = zero;
#pragma unroll
  for (int kb = 0; kb < 64; kb += 32) {
    v16bf a0 = load_a_frag(T, 64, 0,  kb, lane);
    v16bf a1 = load_a_frag(T, 64, 16, kb, lane);
    v16bf b0 = load_b_frag(T, 64, 0,  kb, lane);   // B[k][n] = T[n][k]
    v16bf b1 = load_b_frag(T, 64, 16, kb, lane);
    c00 = wmma_bf16(a0, b0, c00);
    c10 = wmma_bf16(a1, b0, c10);
    c11 = wmma_bf16(a1, b1, c11);
  }

  bf16* Rb = R + (size_t)b * 416;
  Rb[2 * lane]     = T[2 * lane];
  Rb[2 * lane + 1] = T[2 * lane + 1];
  if (lane == 0) Rb[415] = (bf16)0.f;              // pad column

  const int half = lane >> 4, nc = lane & 15;
#pragma unroll
  for (int r = 0; r < 8; ++r) {                    // tile (0,0): m,n in 0..15
    const int m = half * 8 + r, n = nc;
    if (n < m) Rb[64 + (m * (m - 1)) / 2 + n] = (bf16)c00[r];
  }
#pragma unroll
  for (int r = 0; r < 8; ++r) {                    // tile (1,0): m 16..31, n 0..15
    const int m = 16 + half * 8 + r, n = nc;
    if (m < 27) Rb[64 + (m * (m - 1)) / 2 + n] = (bf16)c10[r];
  }
#pragma unroll
  for (int r = 0; r < 8; ++r) {                    // tile (1,1): m,n 16..31
    const int m = 16 + half * 8 + r, n = 16 + nc;
    if (m < 27 && n < m) Rb[64 + (m * (m - 1)) / 2 + n] = (bf16)c11[r];
  }
}

// ---------------------------------------------------------------------------
// Host-side launcher
// ---------------------------------------------------------------------------
static inline size_t ws_take(size_t& off, size_t bytes) {
  size_t r = off;
  off = off + ((bytes + 255) & ~(size_t)255);
  return r;
}

extern "C" void kernel_launch(void* const* d_in, const int* in_sizes, int n_in,
                              void* d_out, int out_size, void* d_ws, size_t ws_size,
                              hipStream_t stream) {
  (void)in_sizes; (void)n_in; (void)out_size; (void)ws_size;

  const float* dense_x = (const float*)d_in[0];
  const int*   sidx    = (const int*)  d_in[1];
  const float* emb     = (const float*)d_in[2];
  const float* bW0 = (const float*)d_in[3];  const float* bb0 = (const float*)d_in[4];
  const float* bW1 = (const float*)d_in[5];  const float* bb1 = (const float*)d_in[6];
  const float* bW2 = (const float*)d_in[7];  const float* bb2 = (const float*)d_in[8];
  const float* tW0 = (const float*)d_in[9];  const float* tb0 = (const float*)d_in[10];
  const float* tW1 = (const float*)d_in[11]; const float* tb1 = (const float*)d_in[12];
  const float* tW2 = (const float*)d_in[13]; const float* tb2 = (const float*)d_in[14];
  float* out = (float*)d_out;

  char* ws = (char*)d_ws;
  size_t off = 0;
  bf16* xb   = (bf16*)(ws + ws_take(off, (size_t)DLRM_B * 32  * 2)); // dense pad 13->32
  bf16* w0b  = (bf16*)(ws + ws_take(off, (size_t)512 * 32  * 2));
  bf16* w1b  = (bf16*)(ws + ws_take(off, (size_t)256 * 512 * 2));
  bf16* w2b  = (bf16*)(ws + ws_take(off, (size_t)64  * 256 * 2));
  bf16* tw0b = (bf16*)(ws + ws_take(off, (size_t)512 * 416 * 2)); // pad 415->416
  bf16* tw1b = (bf16*)(ws + ws_take(off, (size_t)256 * 512 * 2));
  bf16* tw2b = (bf16*)(ws + ws_take(off, (size_t)16  * 256 * 2)); // pad N 1->16
  bf16* h1   = (bf16*)(ws + ws_take(off, (size_t)DLRM_B * 512 * 2)); // reused as t1
  bf16* h2   = (bf16*)(ws + ws_take(off, (size_t)DLRM_B * 256 * 2)); // reused as t2
  bf16* h3   = (bf16*)(ws + ws_take(off, (size_t)DLRM_B * 64  * 2));
  bf16* Rbuf = (bf16*)(ws + ws_take(off, (size_t)DLRM_B * 416 * 2));

  auto cvt = [&](const float* src, bf16* dst, int N, int K, int Np, int Kp) {
    int total = Np * Kp;
    cvt_pad_bf16<<<(total + 255) / 256, 256, 0, stream>>>(src, dst, N, K, Np, Kp);
  };
  cvt(dense_x, xb,  DLRM_B, 13,  DLRM_B, 32);
  cvt(bW0, w0b,  512, 13,  512, 32);
  cvt(bW1, w1b,  256, 512, 256, 512);
  cvt(bW2, w2b,  64,  256, 64,  256);
  cvt(tW0, tw0b, 512, 415, 512, 416);
  cvt(tW1, tw1b, 256, 512, 256, 512);
  cvt(tW2, tw2b, 1,   256, 16,  256);

  auto gemm = [&](const bf16* A, const bf16* W, const float* bias, bf16* O,
                  int M, int N, int Kp) {
    int waves = (M / 16) * (N / 64);
    int threads = waves * 32;
    gemm_bias_act<4, true><<<(threads + 255) / 256, 256, 0, stream>>>(
        A, W, bias, O, M, N, Kp);
  };

  // Bottom MLP
  gemm(xb, w0b, bb0, h1, DLRM_B, 512, 32);
  gemm(h1, w1b, bb1, h2, DLRM_B, 256, 512);
  gemm(h2, w2b, bb2, h3, DLRM_B, 64,  256);

  // Fused embedding gather + pairwise interaction + feature assembly
  interact_kernel<<<DLRM_B / 8, 256, 0, stream>>>(h3, emb, sidx, Rbuf);

  // Top MLP (t1/t2 reuse h1/h2 storage)
  gemm(Rbuf, tw0b, tb0, h1, DLRM_B, 512, 416);
  gemm(h1,   tw1b, tb1, h2, DLRM_B, 256, 512);

  // Final layer + sigmoid -> fp32 output [B,1]
  {
    int waves = DLRM_B / 16;
    int threads = waves * 32;
    gemm_final_sigmoid<<<(threads + 255) / 256, 256, 0, stream>>>(
        h2, tw2b, tb2, out, DLRM_B, 256);
  }
}
